// DCC_12833362280503
// MI455X (gfx1250) — matrix operand order repaired and checked by reading, not compiled, not run on hardware
//
#include <hip/hip_runtime.h>

typedef __attribute__((ext_vector_type(16))) _Float16 v16h;
typedef __attribute__((ext_vector_type(8)))  _Float16 v8h;
typedef __attribute__((ext_vector_type(4)))  _Float16 v4h;
typedef __attribute__((ext_vector_type(8)))  float    v8f;
typedef __attribute__((ext_vector_type(4)))  float    v4f;

#define N_ROWS   8192              // B*E = 128*64
#define DIM      128               // feature dim
#define N_TILES  512               // N_ROWS / 16
#define Q_TILES  128               // column tiles per wave (N_TILES / 4)
#define L2E_T    14.426950408889634f  // (1/T) * log2(e) = 10 * 1.442695...

#if __has_builtin(__builtin_amdgcn_exp2f)
#define FAST_EXP2(x) __builtin_amdgcn_exp2f(x)
#else
#define FAST_EXP2(x) exp2f(x)
#endif

union V16 { v16h v; v8h h[2]; };

// ---------------------------------------------------------------------------
// Kernel 1: L2-normalize each 128-float row, emit f16.  One wave32 per row.
// ---------------------------------------------------------------------------
__global__ void dcc_normalize_f16(const float* __restrict__ in,
                                  _Float16* __restrict__ out) {
    const int wave = threadIdx.x >> 5;
    const int lane = threadIdx.x & 31;
    const int row  = blockIdx.x * (blockDim.x >> 5) + wave;

    const float* p = in + (size_t)row * DIM + lane * 4;
    v4f x = *(const v4f*)p;
    float ss = x.x * x.x + x.y * x.y + x.z * x.z + x.w * x.w;
    #pragma unroll
    for (int m = 16; m >= 1; m >>= 1) ss += __shfl_xor(ss, m, 32);
    float r = rsqrtf(ss);

    v4h y;
    y.x = (_Float16)(x.x * r);
    y.y = (_Float16)(x.y * r);
    y.z = (_Float16)(x.z * r);
    y.w = (_Float16)(x.w * r);
    *(v4h*)(out + (size_t)row * DIM + lane * 4) = y;
}

// ---------------------------------------------------------------------------
// Kernel 2: fused Gram (f16 WMMA) + exp2 + partial row sums.
// Wave task = (rowTile 0..511, quarter 0..3).  Two column tiles are processed
// per iteration with independent B buffers / accumulator chains so loads,
// WMMA (XDL) and exp (TRANS) from the two tiles can overlap.
// ---------------------------------------------------------------------------
__global__ void dcc_gram_expsum(const _Float16* __restrict__ f,
                                float* __restrict__ ws_tot,
                                float* __restrict__ ws_pos) {
    const int wave    = threadIdx.x >> 5;
    const int lane    = threadIdx.x & 31;
    const int task    = blockIdx.x * 8 + wave;        // 0..2047
    const int rowTile = task >> 2;                    // 0..511
    const int quarter = task & 3;                     // 0..3
    const int rowBase = rowTile * 16;
    const int lrow    = lane & 15;
    const int hi      = lane >> 4;                    // 0: lanes 0-15, 1: 16-31

    // --- Preload A tile: 16x128 f16 as four 16x32 A chunks, contiguous k order (B matches) -----
    // lanes 0-15: M=lane, halves0-15 = K0-15
    // lanes 16-31: M=lane-16, halves0-15 = K16-31
    const int hselA = hi * 16;
    const _Float16* pa = f + (size_t)(rowBase + lrow) * DIM;
    V16 a[4];
    #pragma unroll
    for (int k = 0; k < 4; ++k) {
        a[k].h[0] = *(const v8h*)(pa + k * 32 + hselA);
        a[k].h[1] = *(const v8h*)(pa + k * 32 + 8 + hselA);
    }

    // B 32x16 layout: lanes 0-15 hold N=lane, K kk+0..15 contiguous;
    //                 lanes 16-31 hold N=lane-16, K kk+16..31.
    const int koffB = hi * 16;

    float tot[8];
    #pragma unroll
    for (int g = 0; g < 8; ++g) tot[g] = 0.0f;

    const int ct0 = quarter * Q_TILES;
    for (int ct = ct0; ct < ct0 + Q_TILES; ct += 2) {
        const _Float16* pb0 = f + (size_t)((ct    ) * 16 + lrow) * DIM;
        const _Float16* pb1 = f + (size_t)((ct + 1) * 16 + lrow) * DIM;

        // Issue ALL 16 b128 loads for both tiles before any WMMA.
        V16 b0[4], b1[4];
        #pragma unroll
        for (int k = 0; k < 4; ++k) {
            b0[k].h[0] = *(const v8h*)(pb0 + k * 32 + koffB);
            b0[k].h[1] = *(const v8h*)(pb0 + k * 32 + koffB + 8);
        }
        #pragma unroll
        for (int k = 0; k < 4; ++k) {
            b1[k].h[0] = *(const v8h*)(pb1 + k * 32 + koffB);
            b1[k].h[1] = *(const v8h*)(pb1 + k * 32 + koffB + 8);
        }

        v8f c0 = {}, c1 = {};
        #pragma unroll
        for (int k = 0; k < 4; ++k)
            c0 = __builtin_amdgcn_wmma_f32_16x16x32_f16(
                     false, a[k].v, false, b0[k].v, (short)0, c0, false, false);
        #pragma unroll
        for (int k = 0; k < 4; ++k)
            c1 = __builtin_amdgcn_wmma_f32_16x16x32_f16(
                     false, a[k].v, false, b1[k].v, (short)0, c1, false, false);

        #pragma unroll
        for (int g = 0; g < 8; ++g)
            tot[g] += FAST_EXP2(c0[g] * L2E_T);
        #pragma unroll
        for (int g = 0; g < 8; ++g)
            tot[g] += FAST_EXP2(c1[g] * L2E_T);
    }

    // --- Reduce tot across the 16-lane half (C layout: N = lane&15) -------
    #pragma unroll
    for (int g = 0; g < 8; ++g) {
        #pragma unroll
        for (int m = 8; m >= 1; m >>= 1)
            tot[g] += __shfl_xor(tot[g], m, 32);
    }
    // Lane 0 holds rows M=0..7; lane 16 holds rows M=8..15.
    if (lrow == 0) {
        #pragma unroll
        for (int g = 0; g < 8; ++g)
            atomicAdd(&ws_tot[rowBase + hi * 8 + g], tot[g]);
    }

    // --- Positive block: 4 aligned column tiles, in exactly one quarter ---
    const int posLo = (rowBase >> 6) << 2;            // first positive tile
    if ((posLo >> 7) == quarter) {                    // wave-uniform branch
        float pos[8];
        #pragma unroll
        for (int g = 0; g < 8; ++g) pos[g] = 0.0f;

        #pragma unroll
        for (int t = 0; t < 4; t += 2) {
            const _Float16* pb0 = f + (size_t)((posLo + t    ) * 16 + lrow) * DIM;
            const _Float16* pb1 = f + (size_t)((posLo + t + 1) * 16 + lrow) * DIM;
            V16 b0[4], b1[4];
            #pragma unroll
            for (int k = 0; k < 4; ++k) {
                b0[k].h[0] = *(const v8h*)(pb0 + k * 32 + koffB);
                b0[k].h[1] = *(const v8h*)(pb0 + k * 32 + koffB + 8);
            }
            #pragma unroll
            for (int k = 0; k < 4; ++k) {
                b1[k].h[0] = *(const v8h*)(pb1 + k * 32 + koffB);
                b1[k].h[1] = *(const v8h*)(pb1 + k * 32 + koffB + 8);
            }
            v8f c0 = {}, c1 = {};
            #pragma unroll
            for (int k = 0; k < 4; ++k)
                c0 = __builtin_amdgcn_wmma_f32_16x16x32_f16(
                         false, a[k].v, false, b0[k].v, (short)0, c0, false, false);
            #pragma unroll
            for (int k = 0; k < 4; ++k)
                c1 = __builtin_amdgcn_wmma_f32_16x16x32_f16(
                         false, a[k].v, false, b1[k].v, (short)0, c1, false, false);
            #pragma unroll
            for (int g = 0; g < 8; ++g)
                pos[g] += FAST_EXP2(c0[g] * L2E_T) + FAST_EXP2(c1[g] * L2E_T);
        }
        #pragma unroll
        for (int g = 0; g < 8; ++g) {
            #pragma unroll
            for (int m = 8; m >= 1; m >>= 1)
                pos[g] += __shfl_xor(pos[g], m, 32);
        }
        if (lrow == 0) {
            #pragma unroll
            for (int g = 0; g < 8; ++g)
                atomicAdd(&ws_pos[rowBase + hi * 8 + g], pos[g]);
        }
    }
}

// ---------------------------------------------------------------------------
// Kernel 3: loss = mean( log(tot) - log(pos) ) over 8192 rows.
// ---------------------------------------------------------------------------
__global__ void dcc_loss_reduce(const float* __restrict__ ws_tot,
                                const float* __restrict__ ws_pos,
                                float* __restrict__ out) {
    const int r    = blockIdx.x * blockDim.x + threadIdx.x;
    const int lane = threadIdx.x & 31;
    float s = __logf(ws_tot[r]) - __logf(ws_pos[r]);
    #pragma unroll
    for (int m = 16; m >= 1; m >>= 1) s += __shfl_xor(s, m, 32);
    if (lane == 0)
        atomicAdd(out, s * (1.0f / (float)N_ROWS));
}

// ---------------------------------------------------------------------------
extern "C" void kernel_launch(void* const* d_in, const int* in_sizes, int n_in,
                              void* d_out, int out_size, void* d_ws, size_t ws_size,
                              hipStream_t stream) {
    const float* feature = (const float*)d_in[0];     // [128,64,128] f32
    float* out = (float*)d_out;                       // scalar loss

    // workspace layout: [0, 2MB) f16 normalized features,
    //                   then tot[8192], pos[8192] f32 accumulators
    _Float16* fh   = (_Float16*)d_ws;
    float* ws_tot  = (float*)((char*)d_ws + (size_t)N_ROWS * DIM * 2);
    float* ws_pos  = ws_tot + N_ROWS;

    hipMemsetAsync(d_out, 0, sizeof(float), stream);
    hipMemsetAsync(ws_tot, 0, 2 * N_ROWS * sizeof(float), stream);

    // 8 rows per 256-thread block -> 1024 blocks
    dcc_normalize_f16<<<N_ROWS / 8, 256, 0, stream>>>(feature, fh);

    // 512 row tiles x 4 column quarters = 2048 waves, 8 per block -> 256 blocks
    dcc_gram_expsum<<<256, 256, 0, stream>>>(fh, ws_tot, ws_pos);

    // 8192 rows, 256 threads per block -> 32 blocks
    dcc_loss_reduce<<<N_ROWS / 256, 256, 0, stream>>>(ws_tot, ws_pos, out);
}
